// RNN_LOCAL_58514634440781
// MI455X (gfx1250) — compile-verified, matrix operand order checked by reading, not run on hardware
//
#include <hip/hip_runtime.h>
#include <hip/hip_bf16.h>
#include <math.h>

// ---------------- problem constants ----------------
#define S_LEN   256
#define BATCH   128
#define DDIM    128
#define HDIM    512
#define H2DIM   1024   // 2H
#define GDIM    1536   // 3H
#define ODIM    25
#define TEMP_F  0.1f

// persistent kernel shape: 32 blocks x 8 waves = 256 waves (co-resident)
#define NBLOCKS  32
#define NTHREADS 256
#define NWAVES   ((NBLOCKS * NTHREADS) / 32)

typedef __attribute__((ext_vector_type(16))) _Float16 v16h;
typedef __attribute__((ext_vector_type(8)))  _Float16 v8h;
typedef __attribute__((ext_vector_type(8)))  float    v8f;

#define WMMA16(a, b, c) \
  __builtin_amdgcn_wmma_f32_16x16x32_f16(false, (a), false, (b), (short)0, (c), false, false)

union AFrag { v16h v; v8h h[2]; };

// A-fragment: lane l holds row m = mt*16 + (l&15); lanes 0-15 K={0..7,16..23},
// lanes 16-31 K={8..15,24..31} of the 32-wide K block. Two 16B loads.
__device__ __forceinline__ v16h load_a_frag(const _Float16* act, int ldk, int mt, int kt, int lane) {
  int m  = mt * 16 + (lane & 15);
  int kb = kt * 32 + (lane >> 4) * 8;
  const _Float16* p = act + (size_t)m * ldk + kb;
  AFrag f;
  f.h[0] = *(const v8h*)(p);
  f.h[1] = *(const v8h*)(p + 16);
  return f.v;
}

// B-fragment: pre-packed so each lane reads 16 contiguous halves (32B, coalesced).
__device__ __forceinline__ v16h load_b_frag(const _Float16* packed, int numNt, int kt, int nt, int lane) {
  return *(const v16h*)(packed + (((size_t)(kt * numNt + nt) * 32 + lane) << 4));
}

// C init = bias[col] broadcast (all 8 acc elements share one output column).
__device__ __forceinline__ v8f bias_acc(const float* bias, int nt, int lane) {
  float b = bias[nt * 16 + (lane & 15)];
  v8f acc;
#pragma unroll
  for (int j = 0; j < 8; ++j) acc[j] = b;
  return acc;
}

// D layout: VGPR j, lanes 0-15 -> row mt*16+j, lanes 16-31 -> row mt*16+8+j; col = nt*16+(l&15)
__device__ __forceinline__ void store_f16_relu(_Float16* out, int ldo, int mt, int nt, int lane, v8f acc) {
  int col = nt * 16 + (lane & 15);
  int rb  = mt * 16 + ((lane >> 4) << 3);
#pragma unroll
  for (int j = 0; j < 8; ++j) {
    float v = acc[j];
    out[(size_t)(rb + j) * ldo + col] = (_Float16)(v > 0.f ? v : 0.f);
  }
}

__device__ __forceinline__ void store_f32_tile(float* out, int ldo, int mt, int nt, int lane, v8f acc) {
  int col = nt * 16 + (lane & 15);
  int rb  = mt * 16 + ((lane >> 4) << 3);
#pragma unroll
  for (int j = 0; j < 8; ++j) out[(size_t)(rb + j) * ldo + col] = acc[j];
}

// branch-free fast transcendentals (v_exp/v_log paths, no libm range-reduction)
__device__ __forceinline__ float sigmoidf_(float x) { return 1.f / (1.f + __expf(-x)); }
__device__ __forceinline__ float tanhf_(float x) {
  float xc = fminf(fmaxf(x, -15.f), 15.f);   // avoid inf/inf
  float e  = __expf(2.f * xc);
  return (e - 1.f) / (e + 1.f);
}

// ---------------- grid barrier (sense-reversing, device scope) ----------------
__device__ __forceinline__ void gbar(unsigned* cnt, unsigned* gen) {
  __syncthreads();
  if (threadIdx.x == 0) {
    __threadfence();
    unsigned g = __hip_atomic_load(gen, __ATOMIC_RELAXED, __HIP_MEMORY_SCOPE_AGENT);
    unsigned v = __hip_atomic_fetch_add(cnt, 1u, __ATOMIC_ACQ_REL, __HIP_MEMORY_SCOPE_AGENT) + 1u;
    if (v == (unsigned)NBLOCKS) {
      __hip_atomic_store(cnt, 0u, __ATOMIC_RELAXED, __HIP_MEMORY_SCOPE_AGENT);
      __hip_atomic_fetch_add(gen, 1u, __ATOMIC_RELEASE, __HIP_MEMORY_SCOPE_AGENT);
    } else {
      while (__hip_atomic_load(gen, __ATOMIC_ACQUIRE, __HIP_MEMORY_SCOPE_AGENT) == g) {
        __builtin_amdgcn_s_sleep(2);
      }
    }
    __threadfence();
  }
  __syncthreads();
}

// ---------------- prologue kernels ----------------
// Pack fp32 [K,N] weight -> f16 B-fragment tiles (lane-major, 16 halves/lane).
__global__ void k_pack(const float* __restrict__ W, _Float16* __restrict__ P, int Kdim, int N) {
  int total = Kdim * N;
  int numNt = N >> 4;
  int stride = gridDim.x * blockDim.x;
  for (int idx = blockIdx.x * blockDim.x + threadIdx.x; idx < total; idx += stride) {
    int e      = idx & 15;
    int lane   = (idx >> 4) & 31;
    int tileid = idx >> 9;
    int kt = tileid / numNt, nt = tileid % numNt;
    int half = lane >> 4;
    int n = (nt << 4) + (lane & 15);
    int k = (kt << 5) + ((e < 8) ? (half * 8 + e) : (16 + half * 8 + (e - 8)));
    P[idx] = (_Float16)W[(size_t)k * N + n];
  }
}

// x [B,S,D] fp32 -> x16 [S*B, D] f16 (scan-major rows)
__global__ void k_xcvt(const float* __restrict__ x, _Float16* __restrict__ x16) {
  int total = S_LEN * BATCH * DDIM;
  int stride = gridDim.x * blockDim.x;
  for (int idx = blockIdx.x * blockDim.x + threadIdx.x; idx < total; idx += stride) {
    int d = idx & (DDIM - 1);
    int r = idx >> 7;            // row = s*B + b
    int b = r & (BATCH - 1);
    int s = r >> 7;
    x16[idx] = (_Float16)x[((size_t)b * S_LEN + s) * DDIM + d];
  }
}

// zero-init carries + barrier state
__global__ void k_init(float* c32, float* h32, _Float16* c16, _Float16* h16,
                       float* nbuf, unsigned* bar) {
  int idx = blockIdx.x * blockDim.x + threadIdx.x;
  int stride = gridDim.x * blockDim.x;
  for (int i = idx; i < BATCH * HDIM; i += stride) {
    c32[i] = 0.f; h32[i] = 0.f;
    c16[i] = (_Float16)0.f; h16[i] = (_Float16)0.f;
  }
  for (int i = idx; i < 2 * BATCH; i += stride) nbuf[i] = 0.f;
  if (idx < 2) bar[idx] = 0u;
}

// one-shot: cin16[s*B+b][h] = tanh(x16 @ w_cin + b_cin), WMMA f16, 2-wide N blocking
__global__ void k_cin(const _Float16* __restrict__ x16, const _Float16* __restrict__ wcp,
                      const float* __restrict__ b_cin, _Float16* __restrict__ cin16) {
  const int lane = threadIdx.x & 31;
  const int wave = (blockIdx.x * blockDim.x + threadIdx.x) >> 5;
  const int nw   = (gridDim.x * blockDim.x) >> 5;
  const int Mt  = (S_LEN * BATCH) / 16;  // 2048
  const int Nt  = HDIM / 16;             // 32
  const int NtP = Nt / 2;                // 16 pairs per row
  for (int tp = wave; tp < Mt * NtP; tp += nw) {
    int mt = tp / NtP, nt0 = (tp % NtP) * 2;
    v8f acc0 = bias_acc(b_cin, nt0, lane);
    v8f acc1 = bias_acc(b_cin, nt0 + 1, lane);
    for (int kt = 0; kt < DDIM / 32; ++kt) {
      v16h a  = load_a_frag(x16, DDIM, mt, kt, lane);
      v16h b0 = load_b_frag(wcp, Nt, kt, nt0, lane);
      v16h b1 = load_b_frag(wcp, Nt, kt, nt0 + 1, lane);
      acc0 = WMMA16(a, b0, acc0);
      acc1 = WMMA16(a, b1, acc1);
    }
    int rb = mt * 16 + ((lane >> 4) << 3);
    int c0 = nt0 * 16 + (lane & 15);
#pragma unroll
    for (int j = 0; j < 8; ++j) {
      cin16[(size_t)(rb + j) * HDIM + c0]      = (_Float16)tanhf_(acc0[j]);
      cin16[(size_t)(rb + j) * HDIM + c0 + 16] = (_Float16)tanhf_(acc1[j]);
    }
  }
}

// ---------------- persistent scan kernel ----------------
__global__ void __launch_bounds__(NTHREADS)
k_main(const _Float16* __restrict__ cin16,
       const _Float16* __restrict__ w1p, const _Float16* __restrict__ w2p,
       const _Float16* __restrict__ w3p,
       const _Float16* __restrict__ wihp, const _Float16* __restrict__ whhp,
       const float* __restrict__ b1, const float* __restrict__ b2, const float* __restrict__ b3,
       const float* __restrict__ w4, const float* __restrict__ b4,
       const float* __restrict__ b_ih, const float* __restrict__ b_hh,
       const float* __restrict__ noise_u,
       _Float16* __restrict__ z1, _Float16* __restrict__ z2, _Float16* __restrict__ z3,
       float* __restrict__ gx, float* __restrict__ gh,
       float* __restrict__ c32, float* __restrict__ h32,
       _Float16* __restrict__ c16, _Float16* __restrict__ h16,
       float* __restrict__ nbuf, float* __restrict__ hf32,
       const float* __restrict__ w_out, const float* __restrict__ b_out,
       float* __restrict__ out, unsigned* __restrict__ bar) {
  const int lane = threadIdx.x & 31;
  const int wave = (blockIdx.x * blockDim.x + threadIdx.x) >> 5;
  const int tid  = blockIdx.x * blockDim.x + threadIdx.x;
  const int nthr = NBLOCKS * NTHREADS;
  unsigned* cnt = bar;
  unsigned* gen = bar + 1;
  __shared__ float s_alpha[BATCH];

  for (int t = 0; t < S_LEN; ++t) {
    const _Float16* cin_t = cin16 + (size_t)t * BATCH * HDIM;

    // ---- P1: z1 pairs (256) ; gx pairs (384) ; gh pairs (384)  => 1024 units, 4/wave
    for (int u = wave; u < 1024; u += NWAVES) {
      if (u < 256) {                       // z1 = relu([c|cin]@w1+b1), 8mt x 32 pairs
        int mt = u >> 5, nt0 = (u & 31) * 2;
        v8f acc0 = bias_acc(b1, nt0, lane);
        v8f acc1 = bias_acc(b1, nt0 + 1, lane);
        for (int kt = 0; kt < 16; ++kt) {  // k<512 : A = c
          v16h a  = load_a_frag(c16, HDIM, mt, kt, lane);
          v16h b0 = load_b_frag(w1p, 64, kt, nt0, lane);
          v16h b1f = load_b_frag(w1p, 64, kt, nt0 + 1, lane);
          acc0 = WMMA16(a, b0, acc0);
          acc1 = WMMA16(a, b1f, acc1);
        }
        for (int kt = 16; kt < 32; ++kt) { // k>=512 : A = c_in[t]
          v16h a  = load_a_frag(cin_t, HDIM, mt, kt - 16, lane);
          v16h b0 = load_b_frag(w1p, 64, kt, nt0, lane);
          v16h b1f = load_b_frag(w1p, 64, kt, nt0 + 1, lane);
          acc0 = WMMA16(a, b0, acc0);
          acc1 = WMMA16(a, b1f, acc1);
        }
        store_f16_relu(z1, H2DIM, mt, nt0, lane, acc0);
        store_f16_relu(z1, H2DIM, mt, nt0 + 1, lane, acc1);
      } else if (u < 640) {                // gx = c@w_ih+b_ih, 8mt x 48 pairs
        int tt = u - 256;
        int mt = tt / 48, nt0 = (tt % 48) * 2;
        v8f acc0 = bias_acc(b_ih, nt0, lane);
        v8f acc1 = bias_acc(b_ih, nt0 + 1, lane);
        for (int kt = 0; kt < 16; ++kt) {
          v16h a  = load_a_frag(c16, HDIM, mt, kt, lane);
          v16h b0 = load_b_frag(wihp, 96, kt, nt0, lane);
          v16h b1f = load_b_frag(wihp, 96, kt, nt0 + 1, lane);
          acc0 = WMMA16(a, b0, acc0);
          acc1 = WMMA16(a, b1f, acc1);
        }
        store_f32_tile(gx, GDIM, mt, nt0, lane, acc0);
        store_f32_tile(gx, GDIM, mt, nt0 + 1, lane, acc1);
      } else {                             // gh = h@w_hh+b_hh
        int tt = u - 640;
        int mt = tt / 48, nt0 = (tt % 48) * 2;
        v8f acc0 = bias_acc(b_hh, nt0, lane);
        v8f acc1 = bias_acc(b_hh, nt0 + 1, lane);
        for (int kt = 0; kt < 16; ++kt) {
          v16h a  = load_a_frag(h16, HDIM, mt, kt, lane);
          v16h b0 = load_b_frag(whhp, 96, kt, nt0, lane);
          v16h b1f = load_b_frag(whhp, 96, kt, nt0 + 1, lane);
          acc0 = WMMA16(a, b0, acc0);
          acc1 = WMMA16(a, b1f, acc1);
        }
        store_f32_tile(gh, GDIM, mt, nt0, lane, acc0);
        store_f32_tile(gh, GDIM, mt, nt0 + 1, lane, acc1);
      }
    }
    gbar(cnt, gen);

    // ---- P2: z2 = relu(z1@w2+b2), 256 pairs (1/wave)
    for (int u = wave; u < 256; u += NWAVES) {
      int mt = u >> 5, nt0 = (u & 31) * 2;
      v8f acc0 = bias_acc(b2, nt0, lane);
      v8f acc1 = bias_acc(b2, nt0 + 1, lane);
      for (int kt = 0; kt < 32; ++kt) {
        v16h a  = load_a_frag(z1, H2DIM, mt, kt, lane);
        v16h b0 = load_b_frag(w2p, 64, kt, nt0, lane);
        v16h b1f = load_b_frag(w2p, 64, kt, nt0 + 1, lane);
        acc0 = WMMA16(a, b0, acc0);
        acc1 = WMMA16(a, b1f, acc1);
      }
      store_f16_relu(z2, H2DIM, mt, nt0, lane, acc0);
      store_f16_relu(z2, H2DIM, mt, nt0 + 1, lane, acc1);
    }
    gbar(cnt, gen);

    // ---- P3: z3 = relu(z2@w3+b3)
    for (int u = wave; u < 256; u += NWAVES) {
      int mt = u >> 5, nt0 = (u & 31) * 2;
      v8f acc0 = bias_acc(b3, nt0, lane);
      v8f acc1 = bias_acc(b3, nt0 + 1, lane);
      for (int kt = 0; kt < 32; ++kt) {
        v16h a  = load_a_frag(z2, H2DIM, mt, kt, lane);
        v16h b0 = load_b_frag(w3p, 64, kt, nt0, lane);
        v16h b1f = load_b_frag(w3p, 64, kt, nt0 + 1, lane);
        acc0 = WMMA16(a, b0, acc0);
        acc1 = WMMA16(a, b1f, acc1);
      }
      store_f16_relu(z3, H2DIM, mt, nt0, lane, acc0);
      store_f16_relu(z3, H2DIM, mt, nt0 + 1, lane, acc1);
    }
    gbar(cnt, gen);

    // ---- P4: alpha (redundant per block, vectorized dot) + elementwise state update
    if (threadIdx.x < BATCH) {
      int b = threadIdx.x;
      const v8h* zr = (const v8h*)(z3 + (size_t)b * H2DIM);
      float acc = 0.f;
      for (int k8 = 0; k8 < H2DIM / 8; ++k8) {
        v8h zv = zr[k8];
#pragma unroll
        for (int e = 0; e < 8; ++e) acc += (float)zv[e] * w4[k8 * 8 + e];
      }
      float logit = acc + b4[0];
      float u = noise_u[t * BATCH + b];
      float logistic = __logf(u) - __logf(1.f - u);   // u clipped to [1e-6, 1-1e-6]
      s_alpha[b] = sigmoidf_((logit + logistic) * (1.0f / TEMP_F));
    }
    __syncthreads();

    const float* n_in  = nbuf + (t & 1) * BATCH;
    float*       n_out = nbuf + ((t + 1) & 1) * BATCH;
    for (int idx = tid; idx < BATCH * HDIM; idx += nthr) {
      int b = idx >> 9, j = idx & (HDIM - 1);
      float alpha = s_alpha[b];
      const float* gxb = gx + (size_t)b * GDIM;
      const float* ghb = gh + (size_t)b * GDIM;
      float r  = sigmoidf_(gxb[j] + ghb[j]);
      float zg = sigmoidf_(gxb[HDIM + j] + ghb[HDIM + j]);
      float ng = tanhf_(gxb[2 * HDIM + j] + r * ghb[2 * HDIM + j]);
      float h_old = h32[idx];
      float h_gru = (1.f - zg) * ng + zg * h_old;
      float h_new = h_old * (1.f - alpha) + alpha * h_gru;
      float c_old = c32[idx];
      float n_old = n_in[b];
      float new_n = n_old * (1.f - alpha) + 1.f;
      float c_new = (c_old * n_old * (1.f - alpha) + (float)cin_t[idx]) / new_n;
      c32[idx] = c_new; h32[idx] = h_new;
      c16[idx] = (_Float16)c_new; h16[idx] = (_Float16)h_new;
      if (j == 0) n_out[b] = new_n;
    }
    gbar(cnt, gen);
  }

  // ---- F1: final GRU GEMMs (768 pairs, 3/wave)
  for (int u = wave; u < 768; u += NWAVES) {
    if (u < 384) {
      int mt = u / 48, nt0 = (u % 48) * 2;
      v8f acc0 = bias_acc(b_ih, nt0, lane);
      v8f acc1 = bias_acc(b_ih, nt0 + 1, lane);
      for (int kt = 0; kt < 16; ++kt) {
        v16h a  = load_a_frag(c16, HDIM, mt, kt, lane);
        v16h b0 = load_b_frag(wihp, 96, kt, nt0, lane);
        v16h b1f = load_b_frag(wihp, 96, kt, nt0 + 1, lane);
        acc0 = WMMA16(a, b0, acc0);
        acc1 = WMMA16(a, b1f, acc1);
      }
      store_f32_tile(gx, GDIM, mt, nt0, lane, acc0);
      store_f32_tile(gx, GDIM, mt, nt0 + 1, lane, acc1);
    } else {
      int tt = u - 384;
      int mt = tt / 48, nt0 = (tt % 48) * 2;
      v8f acc0 = bias_acc(b_hh, nt0, lane);
      v8f acc1 = bias_acc(b_hh, nt0 + 1, lane);
      for (int kt = 0; kt < 16; ++kt) {
        v16h a  = load_a_frag(h16, HDIM, mt, kt, lane);
        v16h b0 = load_b_frag(whhp, 96, kt, nt0, lane);
        v16h b1f = load_b_frag(whhp, 96, kt, nt0 + 1, lane);
        acc0 = WMMA16(a, b0, acc0);
        acc1 = WMMA16(a, b1f, acc1);
      }
      store_f32_tile(gh, GDIM, mt, nt0, lane, acc0);
      store_f32_tile(gh, GDIM, mt, nt0 + 1, lane, acc1);
    }
  }
  gbar(cnt, gen);

  // ---- F2: h_final = gru_cell elementwise (fp32)
  for (int idx = tid; idx < BATCH * HDIM; idx += nthr) {
    int b = idx >> 9, j = idx & (HDIM - 1);
    const float* gxb = gx + (size_t)b * GDIM;
    const float* ghb = gh + (size_t)b * GDIM;
    float r  = sigmoidf_(gxb[j] + ghb[j]);
    float zg = sigmoidf_(gxb[HDIM + j] + ghb[HDIM + j]);
    float ng = tanhf_(gxb[2 * HDIM + j] + r * ghb[2 * HDIM + j]);
    hf32[idx] = (1.f - zg) * ng + zg * h32[idx];
  }
  gbar(cnt, gen);

  // ---- F3: out = h_final @ w_out + b_out  (128x25, O=25 not WMMA-friendly -> fp32 dots)
  for (int oi = tid; oi < BATCH * ODIM; oi += nthr) {
    int b = oi / ODIM, o = oi % ODIM;
    const float* hb = hf32 + (size_t)b * HDIM;
    float acc = b_out[o];
    for (int k = 0; k < HDIM; ++k) acc += hb[k] * w_out[(size_t)k * ODIM + o];
    out[oi] = acc;
  }
}

// ---------------- host launcher ----------------
static inline size_t al256(size_t x) { return (x + 255) & ~(size_t)255; }

extern "C" void kernel_launch(void* const* d_in, const int* in_sizes, int n_in,
                              void* d_out, int out_size, void* d_ws, size_t ws_size,
                              hipStream_t stream) {
  (void)in_sizes; (void)n_in; (void)out_size; (void)ws_size;
  const float* x       = (const float*)d_in[0];
  const float* noise_u = (const float*)d_in[1];
  const float* w_cin   = (const float*)d_in[2];
  const float* b_cin   = (const float*)d_in[3];
  const float* w1 = (const float*)d_in[4];  const float* b1 = (const float*)d_in[5];
  const float* w2 = (const float*)d_in[6];  const float* b2 = (const float*)d_in[7];
  const float* w3 = (const float*)d_in[8];  const float* b3 = (const float*)d_in[9];
  const float* w4 = (const float*)d_in[10]; const float* b4 = (const float*)d_in[11];
  const float* w_ih = (const float*)d_in[12]; const float* b_ih = (const float*)d_in[13];
  const float* w_hh = (const float*)d_in[14]; const float* b_hh = (const float*)d_in[15];
  const float* w_out = (const float*)d_in[16]; const float* b_out = (const float*)d_in[17];
  float* out = (float*)d_out;

  char* p = (char*)d_ws;
  auto take = [&](size_t bytes) { char* r = p; p += al256(bytes); return r; };

  unsigned*  bar   = (unsigned*)take(256);
  _Float16*  x16   = (_Float16*)take((size_t)S_LEN * BATCH * DDIM * 2);
  _Float16*  cin16 = (_Float16*)take((size_t)S_LEN * BATCH * HDIM * 2);
  _Float16*  wcp   = (_Float16*)take((size_t)DDIM * HDIM * 2);
  _Float16*  w1p   = (_Float16*)take((size_t)H2DIM * H2DIM * 2);
  _Float16*  w2p   = (_Float16*)take((size_t)H2DIM * H2DIM * 2);
  _Float16*  w3p   = (_Float16*)take((size_t)H2DIM * H2DIM * 2);
  _Float16*  wihp  = (_Float16*)take((size_t)HDIM * GDIM * 2);
  _Float16*  whhp  = (_Float16*)take((size_t)HDIM * GDIM * 2);
  _Float16*  z1    = (_Float16*)take((size_t)BATCH * H2DIM * 2);
  _Float16*  z2    = (_Float16*)take((size_t)BATCH * H2DIM * 2);
  _Float16*  z3    = (_Float16*)take((size_t)BATCH * H2DIM * 2);
  float*     gx    = (float*)take((size_t)BATCH * GDIM * 4);
  float*     gh    = (float*)take((size_t)BATCH * GDIM * 4);
  float*     c32   = (float*)take((size_t)BATCH * HDIM * 4);
  float*     h32   = (float*)take((size_t)BATCH * HDIM * 4);
  float*     hf32  = (float*)take((size_t)BATCH * HDIM * 4);
  _Float16*  c16   = (_Float16*)take((size_t)BATCH * HDIM * 2);
  _Float16*  h16   = (_Float16*)take((size_t)BATCH * HDIM * 2);
  float*     nbuf  = (float*)take((size_t)2 * BATCH * 4);

  k_init<<<256, 256, 0, stream>>>(c32, h32, c16, h16, nbuf, bar);

  k_pack<<<1024, 256, 0, stream>>>(w1, w1p, H2DIM, H2DIM);
  k_pack<<<1024, 256, 0, stream>>>(w2, w2p, H2DIM, H2DIM);
  k_pack<<<1024, 256, 0, stream>>>(w3, w3p, H2DIM, H2DIM);
  k_pack<<<1024, 256, 0, stream>>>(w_ih, wihp, HDIM, GDIM);
  k_pack<<<1024, 256, 0, stream>>>(w_hh, whhp, HDIM, GDIM);
  k_pack<<<256, 256, 0, stream>>>(w_cin, wcp, DDIM, HDIM);

  k_xcvt<<<1024, 256, 0, stream>>>(x, x16);
  k_cin<<<1024, 256, 0, stream>>>(x16, wcp, b_cin, cin16);

  k_main<<<NBLOCKS, NTHREADS, 0, stream>>>(
      cin16, w1p, w2p, w3p, wihp, whhp, b1, b2, b3, w4, b4, b_ih, b_hh,
      noise_u, z1, z2, z3, gx, gh, c32, h32, c16, h16, nbuf, hf32,
      w_out, b_out, out, bar);
}
